// NLIHead_66417374265586
// MI455X (gfx1250) — matrix operand order repaired and checked by reading, not compiled, not run on hardware
//
#include <hip/hip_runtime.h>
#include <hip/hip_bf16.h>
#include <stdint.h>

// ---------------------------------------------------------------------------
// Problem constants
// ---------------------------------------------------------------------------
#define B_ROWS 16384
#define D_DIM  1024
#define FDIM   1034        // D + 10
#define K_PAD  1056        // 33 * 32  (K padded for 16x16x32 bf16 WMMA)
#define N_PAD  1088        // 17 * 64  (N padded so each wave owns a 32x64 tile)
#define EPSN   1e-12f

typedef __attribute__((ext_vector_type(16))) __bf16 bf16x16;
typedef __attribute__((ext_vector_type(8)))  float  f32x8;

static __device__ __forceinline__ unsigned short f2bf(float f) {
    unsigned u = __float_as_uint(f);
    u += 0x7FFFu + ((u >> 16) & 1u);          // round-to-nearest-even
    return (unsigned short)(u >> 16);
}
static __device__ __forceinline__ float bf2f(unsigned short h) {
    return __uint_as_float(((unsigned)h) << 16);
}
static __device__ __forceinline__ float wred(float x) {
#pragma unroll
    for (int o = 16; o > 0; o >>= 1) x += __shfl_xor(x, o, 32);  // wave32
    return x;
}

// ---------------------------------------------------------------------------
// Kernel 1: pack w1 (fp32 [1034][1034]) -> bf16 [N_PAD][K_PAD], pad b1
// ---------------------------------------------------------------------------
__global__ void pack_w1_kernel(const float* __restrict__ w1,
                               const float* __restrict__ b1,
                               unsigned short* __restrict__ w1b,
                               float* __restrict__ b1p) {
    int idx = blockIdx.x * blockDim.x + threadIdx.x;
    const int total = N_PAD * K_PAD;
    if (idx < total) {
        int n = idx / K_PAD;
        int k = idx - n * K_PAD;
        float v = (n < FDIM && k < FDIM) ? w1[(size_t)n * FDIM + k] : 0.0f;
        w1b[idx] = f2bf(v);
    }
    if (idx < N_PAD) b1p[idx] = (idx < FDIM) ? b1[idx] : 0.0f;
}

// ---------------------------------------------------------------------------
// Kernel 2: per-row feature construction (7 fused reductions), bf16 output row
// feat[row] = [ bf16(h_final[0..1023]), 10 scalar features, 22 zeros ]
// ---------------------------------------------------------------------------
__global__ __launch_bounds__(256)
void feat_kernel(const float* __restrict__ hf, const float* __restrict__ vp,
                 const float* __restrict__ vh, const float* __restrict__ nd,
                 const float* __restrict__ nscale,
                 unsigned short* __restrict__ feat) {
    const int row = blockIdx.x;
    const int tid = threadIdx.x;
    const float* hr = hf + (size_t)row * D_DIM;
    const float* pr = vp + (size_t)row * D_DIM;
    const float* qr = vh + (size_t)row * D_DIM;

    float s_ph = 0.f, s_pp = 0.f, s_hh = 0.f, s_ff = 0.f;
    float s_pn = 0.f, s_hn = 0.f, s_nn = 0.f;
    for (int k = tid; k < D_DIM; k += 256) {
        float p = pr[k], h = qr[k], f = hr[k], n = nd[k];
        s_ph += p * h; s_pp += p * p; s_hh += h * h; s_ff += f * f;
        s_pn += p * n; s_hn += h * n; s_nn += n * n;
    }
    s_ph = wred(s_ph); s_pp = wred(s_pp); s_hh = wred(s_hh); s_ff = wred(s_ff);
    s_pn = wred(s_pn); s_hn = wred(s_hn); s_nn = wred(s_nn);

    __shared__ float lds[8][7];
    const int lane = tid & 31, wave = tid >> 5;
    if (lane == 0) {
        lds[wave][0] = s_ph; lds[wave][1] = s_pp; lds[wave][2] = s_hh;
        lds[wave][3] = s_ff; lds[wave][4] = s_pn; lds[wave][5] = s_hn;
        lds[wave][6] = s_nn;
    }
    __syncthreads();

    unsigned short* fr = feat + (size_t)row * K_PAD;
    if (tid == 0) {
        float t[7] = {0.f, 0.f, 0.f, 0.f, 0.f, 0.f, 0.f};
        for (int w = 0; w < 8; ++w)
            for (int j = 0; j < 7; ++j) t[j] += lds[w][j];
        const float ph = t[0], pp = t[1], hh = t[2], ff = t[3];
        const float pn = t[4], hn = t[5], nn = t[6];
        const float scale = nscale[0];

        const float r_p = sqrtf(pp), r_h = sqrtf(hh), r_f = sqrtf(ff);
        const float inv_p  = 1.0f / fmaxf(r_p, EPSN);
        const float inv_h  = 1.0f / fmaxf(r_h, EPSN);
        const float inv_nd = 1.0f / fmaxf(sqrtf(nn), EPSN);

        const float align      = ph * inv_p * inv_h;
        const float opp        = -align;
        const float energy     = (1.0f + align) * 0.5f;     // K_O == 1.0
        const float expose_neg = (1.0f - align) * 0.5f;
        const float dist_ph    = sqrtf(fmaxf(pp + hh - 2.0f * ph, 0.0f));
        const float anp        = scale * pn * inv_p * inv_nd;
        const float anh        = scale * hn * inv_h * inv_nd;

        fr[D_DIM + 0] = f2bf(align);
        fr[D_DIM + 1] = f2bf(opp);
        fr[D_DIM + 2] = f2bf(energy);
        fr[D_DIM + 3] = f2bf(expose_neg);
        fr[D_DIM + 4] = f2bf(dist_ph);
        fr[D_DIM + 5] = f2bf(r_p);
        fr[D_DIM + 6] = f2bf(r_h);
        fr[D_DIM + 7] = f2bf(r_f);
        fr[D_DIM + 8] = f2bf(anp);
        fr[D_DIM + 9] = f2bf(anh);
        for (int k = FDIM; k < K_PAD; ++k) fr[k] = 0;       // zero K-pad
    }
    for (int k = tid; k < D_DIM; k += 256) fr[k] = f2bf(hr[k]);
}

// ---------------------------------------------------------------------------
// Kernel 3: hidden = relu(feat @ w1^T + b1) via v_wmma_f32_16x16x32_bf16
// Block = 256 threads = 8 waves; wave w owns rows [m0, m0+32) x cols [n0, n0+64):
// per K-step: 12 batched b128 loads (2 A-frags + 4 B-frags in distinct regs)
// feed 8 WMMAs, letting the scheduler overlap tail loads with early WMMAs.
// A-frag (16x32 bf16): lanes 0-15 = row M=lane, K {0..7,16..23};
//                      lanes 16-31 = row M=lane-16, K {8..15,24..31}
// B-frag (32x16 bf16): lane%16 = column, half-wave selects K {0..7,16..23} / {8..15,24..31}
// C layout: VGPR r -> row m0 + r + 8*(lane>=16), col n0 + lane%16
// ---------------------------------------------------------------------------
__global__ __launch_bounds__(256)
void gemm1_wmma_kernel(const unsigned short* __restrict__ feat,
                       const unsigned short* __restrict__ w1b,
                       const float* __restrict__ b1p,
                       unsigned short* __restrict__ hidden) {
    const int lane = threadIdx.x & 31;
    const int wave = threadIdx.x >> 5;
    const int m0   = blockIdx.x * 256 + wave * 32;   // 32 rows per wave
    const int n0   = blockIdx.y * 64;                // 64 cols per wave
    const int half = lane >> 4;      // 0 | 1
    const int l15  = lane & 15;

    union Frag { uint4 q[2]; bf16x16 v; };

    f32x8 acc[2][4] = {};

    const unsigned short* aBase0 = feat + (size_t)(m0 + l15) * K_PAD + half * 8;
    const unsigned short* aBase1 = aBase0 + (size_t)16 * K_PAD;
    const unsigned short* bBase  = w1b + (size_t)(n0 + l15) * K_PAD + half * 8;

    for (int k0 = 0; k0 < K_PAD; k0 += 32) {
        // prefetch the DRAM-resident A stream a few K-steps ahead
        __builtin_prefetch(aBase0 + k0 + 96, 0, 1);
        __builtin_prefetch(aBase1 + k0 + 96, 0, 1);

        // ---- batch all 12 fragment loads into distinct registers ----
        Frag a0, a1, b[4];
        a0.q[0] = *(const uint4*)(aBase0 + k0);
        a0.q[1] = *(const uint4*)(aBase0 + k0 + 16);
        a1.q[0] = *(const uint4*)(aBase1 + k0);
        a1.q[1] = *(const uint4*)(aBase1 + k0 + 16);
#pragma unroll
        for (int t = 0; t < 4; ++t) {
            const unsigned short* bp = bBase + (size_t)(t * 16) * K_PAD + k0;
            b[t].q[0] = *(const uint4*)(bp);
            b[t].q[1] = *(const uint4*)(bp + 16);
        }
        // ---- 8 WMMAs consuming the batch ----
#pragma unroll
        for (int t = 0; t < 4; ++t) {
            acc[0][t] = __builtin_amdgcn_wmma_f32_16x16x32_bf16(
                false, a0.v, false, b[t].v, (short)0, acc[0][t], false, false);
            acc[1][t] = __builtin_amdgcn_wmma_f32_16x16x32_bf16(
                false, a1.v, false, b[t].v, (short)0, acc[1][t], false, false);
        }
    }

    // epilogue: bias + relu, store bf16
#pragma unroll
    for (int m = 0; m < 2; ++m) {
#pragma unroll
        for (int t = 0; t < 4; ++t) {
            const int col = n0 + t * 16 + l15;
            const float bias = b1p[col];
#pragma unroll
            for (int r = 0; r < 8; ++r) {
                const int rowm = m0 + m * 16 + r + half * 8;
                float v = acc[m][t][r] + bias;
                v = v > 0.0f ? v : 0.0f;
                hidden[(size_t)rowm * N_PAD + col] = f2bf(v);
            }
        }
    }
}

// ---------------------------------------------------------------------------
// Kernel 4: logits = hidden @ w2^T + b2   (N=3; bandwidth-bound reduction)
// ---------------------------------------------------------------------------
__global__ __launch_bounds__(256)
void gemm2_kernel(const unsigned short* __restrict__ hidden,
                  const float* __restrict__ w2, const float* __restrict__ b2,
                  float* __restrict__ out) {
    const int row = blockIdx.x;
    const int tid = threadIdx.x;
    const unsigned short* hr = hidden + (size_t)row * N_PAD;

    float a0 = 0.f, a1 = 0.f, a2 = 0.f;
    for (int k = tid; k < FDIM; k += 256) {
        float h = bf2f(hr[k]);
        a0 += h * w2[k];
        a1 += h * w2[FDIM + k];
        a2 += h * w2[2 * FDIM + k];
    }
    a0 = wred(a0); a1 = wred(a1); a2 = wred(a2);

    __shared__ float lds[8][3];
    const int lane = tid & 31, wave = tid >> 5;
    if (lane == 0) { lds[wave][0] = a0; lds[wave][1] = a1; lds[wave][2] = a2; }
    __syncthreads();
    if (tid == 0) {
        float t0 = 0.f, t1 = 0.f, t2 = 0.f;
        for (int w = 0; w < 8; ++w) { t0 += lds[w][0]; t1 += lds[w][1]; t2 += lds[w][2]; }
        out[(size_t)row * 3 + 0] = t0 + b2[0];
        out[(size_t)row * 3 + 1] = t1 + b2[1];
        out[(size_t)row * 3 + 2] = t2 + b2[2];
    }
}

// ---------------------------------------------------------------------------
// Launch
// ---------------------------------------------------------------------------
extern "C" void kernel_launch(void* const* d_in, const int* in_sizes, int n_in,
                              void* d_out, int out_size, void* d_ws, size_t ws_size,
                              hipStream_t stream) {
    const float* h_final = (const float*)d_in[0];
    const float* v_p     = (const float*)d_in[1];
    const float* v_h     = (const float*)d_in[2];
    const float* nd      = (const float*)d_in[3];
    const float* nscale  = (const float*)d_in[4];
    const float* w1      = (const float*)d_in[5];
    const float* b1      = (const float*)d_in[6];
    const float* w2      = (const float*)d_in[7];
    const float* b2      = (const float*)d_in[8];
    float* out = (float*)d_out;

    // workspace carve-out (256B aligned)
    char* ws = (char*)d_ws;
    size_t off = 0;
    auto carve = [&](size_t bytes) -> void* {
        void* p = ws + off;
        off = (off + bytes + 255) & ~(size_t)255;
        return p;
    };
    unsigned short* feat   = (unsigned short*)carve((size_t)B_ROWS * K_PAD * 2); // 34.6 MB
    unsigned short* w1b    = (unsigned short*)carve((size_t)N_PAD * K_PAD * 2);  //  2.3 MB
    float*          b1p    = (float*)carve((size_t)N_PAD * 4);
    unsigned short* hidden = (unsigned short*)carve((size_t)B_ROWS * N_PAD * 2); // 35.7 MB

    // 1) pack w1/b1 to bf16 padded
    {
        int total = N_PAD * K_PAD;
        pack_w1_kernel<<<(total + 255) / 256, 256, 0, stream>>>(w1, b1, w1b, b1p);
    }
    // 2) per-row features
    feat_kernel<<<B_ROWS, 256, 0, stream>>>(h_final, v_p, v_h, nd, nscale, feat);
    // 3) main GEMM: [16384 x 1056] x [1056 x 1088] via bf16 WMMA
    {
        dim3 grid(B_ROWS / 256, N_PAD / 64);
        gemm1_wmma_kernel<<<grid, 256, 0, stream>>>(feat, w1b, b1p, hidden);
    }
    // 4) 3-class projection
    gemm2_kernel<<<B_ROWS, 256, 0, stream>>>(hidden, w2, b2, out);
}
